// Multi_head_attention_86045374808390
// MI455X (gfx1250) — matrix operand order repaired and checked by reading, not compiled, not run on hardware
//
#include <hip/hip_runtime.h>

typedef __bf16 bf16;
typedef __attribute__((ext_vector_type(16))) __bf16 v16bf;
typedef __attribute__((ext_vector_type(8)))  __bf16 bf16x8;
typedef __attribute__((ext_vector_type(8)))  float  v8f;

static constexpr int Bb = 8, S = 1024, D = 1024, H = 16, HD = 64;
static constexpr int M_TOK = Bb * S;  // 8192 tokens

// ---------- helpers ----------

__device__ inline bf16 f2bf(float f) {
  unsigned u = __builtin_bit_cast(unsigned, f);
  unsigned r = (u + 0x7fffu + ((u >> 16) & 1u)) >> 16;  // round-to-nearest-even
  unsigned short s = (unsigned short)r;
  return __builtin_bit_cast(bf16, s);
}

union V16U { v16bf v; bf16x8 h[2]; };

__device__ inline v16bf ld_frag(const bf16* p0, const bf16* p1) {
  V16U u;
  u.h[0] = *reinterpret_cast<const bf16x8*>(p0);
  u.h[1] = *reinterpret_cast<const bf16x8*>(p1);
  return u.v;
}

__device__ inline v8f wmma_bf16(v16bf a, v16bf b, v8f c) {
  // (neg_a, A, neg_b, B, c_mod, C, reuse_a, reuse_b)
  return __builtin_amdgcn_wmma_f32_16x16x32_bf16(false, a, false, b, (short)0, c,
                                                 false, false);
}

// CDNA5 async global->LDS copy (16B per lane), tracked by ASYNCcnt.
// Low 32 bits of a generic LDS pointer are the LDS byte offset.
__device__ inline void async_b128(const bf16* g, bf16* l) {
  const unsigned lds = (unsigned)(unsigned long long)l;
  asm volatile("global_load_async_to_lds_b128 %0, %1, off"
               :: "v"(lds), "v"(g)
               : "memory");
}

template<int N>
__device__ inline void async_wait() {
  asm volatile("s_wait_asynccnt %0" :: "n"(N) : "memory");
}

// ---------- one-pass f32 -> bf16 cast (bandwidth bound) ----------

__global__ __launch_bounds__(256)
void cast_f32_bf16(const float* __restrict__ src, bf16* __restrict__ dst, int n) {
  const int i = (blockIdx.x * 256 + threadIdx.x) * 8;
  if (i >= n) return;
  const float4 a = reinterpret_cast<const float4*>(src + i)[0];
  const float4 b = reinterpret_cast<const float4*>(src + i)[1];
  bf16x8 o;
  o[0] = f2bf(a.x); o[1] = f2bf(a.y); o[2] = f2bf(a.z); o[3] = f2bf(a.w);
  o[4] = f2bf(b.x); o[5] = f2bf(b.y); o[6] = f2bf(b.z); o[7] = f2bf(b.w);
  *reinterpret_cast<bf16x8*>(dst + i) = o;
}

// ---------- projection GEMM: Y[M,N] = X[M,K] * W[N,K]^T (all bf16 in) ----------
// Block tile 128x256x32, 256 threads (8 waves, 2x4), each wave owns 64x64 =>
// 4x4 16x16 accumulators, 16 WMMAs per K-step per wave off 8 fragment loads.
// Tiles staged with async global->LDS copies, double buffered on ASYNCcnt.

enum { OUT_BF16 = 0, OUT_VT = 1, OUT_F32 = 2 };

template<int OMODE>
__global__ __launch_bounds__(256)
void gemm_bf16(const bf16* __restrict__ A, const bf16* __restrict__ Wb,
               void* __restrict__ Out) {
  constexpr int BM = 128, BN = 256, BK = 32, LDT = 40, NK = D / BK;
  __shared__ bf16 As[2][BM * LDT];   // 2 x 10 KB
  __shared__ bf16 Bs[2][BN * LDT];   // 2 x 20 KB

  const int tid   = threadIdx.x;
  const int lane  = tid & 31;
  const int wave  = tid >> 5;
  const int waveM = wave & 1;   // 0..1
  const int waveN = wave >> 1;  // 0..3
  const int mblk  = blockIdx.y * BM;
  const int nblk  = blockIdx.x * BN;

  const int mA  = lane & 15;
  const int khA = lane >> 4;

  v8f acc[4][4];
  const v8f vz = {0.f, 0.f, 0.f, 0.f, 0.f, 0.f, 0.f, 0.f};
#pragma unroll
  for (int mi = 0; mi < 4; ++mi)
#pragma unroll
    for (int ni = 0; ni < 4; ++ni) acc[mi][ni] = vz;

  // A tile: 128 rows x 32 cols -> 2 x b128 per thread
  // B tile: 256 rows x 32 cols -> 4 x b128 per thread (one row per thread)
  auto issue = [&](int kt, int buf) {
    const bf16* ga = A + (size_t)(mblk + (tid >> 1)) * D + kt * BK + (tid & 1) * 16;
    bf16* la = &As[buf][(tid >> 1) * LDT + (tid & 1) * 16];
    async_b128(ga,     la);
    async_b128(ga + 8, la + 8);
    const bf16* gb = Wb + (size_t)(nblk + tid) * D + kt * BK;
    bf16* lb = &Bs[buf][tid * LDT];
    async_b128(gb,      lb);
    async_b128(gb + 8,  lb + 8);
    async_b128(gb + 16, lb + 16);
    async_b128(gb + 24, lb + 24);
  };

  issue(0, 0);
  for (int kt = 0; kt < NK; ++kt) {
    const int cur = kt & 1;
    if (kt + 1 < NK) {
      issue(kt + 1, cur ^ 1);
      async_wait<6>();   // previous batch of 6 copies complete
    } else {
      async_wait<0>();
    }
    __syncthreads();     // everyone's copies for tile kt are in LDS

    v16bf bfr[4];
#pragma unroll
    for (int ni = 0; ni < 4; ++ni) {
      const bf16* p = &Bs[cur][(waveN * 64 + ni * 16 + mA) * LDT + khA * 16];
      bfr[ni] = ld_frag(p, p + 8);                                // B layout
    }
#pragma unroll
    for (int mi = 0; mi < 4; ++mi) {
      const bf16* p = &As[cur][(waveM * 64 + mi * 16 + mA) * LDT];
      const v16bf af = ld_frag(p + khA * 16, p + khA * 16 + 8);   // A layout
#pragma unroll
      for (int ni = 0; ni < 4; ++ni)
        acc[mi][ni] = wmma_bf16(af, bfr[ni], acc[mi][ni]);
    }

    __syncthreads();     // reads done before this buffer is refilled
  }

  // --- epilogue: C layout = row r in VGPR r, N across 16 lanes, M+8 in hi half
#pragma unroll
  for (int mi = 0; mi < 4; ++mi)
#pragma unroll
    for (int ni = 0; ni < 4; ++ni)
#pragma unroll
      for (int r = 0; r < 8; ++r) {
        const int gm = mblk + waveM * 64 + mi * 16 + r + 8 * khA;
        const int gn = nblk + waveN * 64 + ni * 16 + mA;
        const float val = acc[mi][ni][r];
        if constexpr (OMODE == OUT_BF16) {
          ((bf16*)Out)[(size_t)gm * D + gn] = f2bf(val);
        } else if constexpr (OMODE == OUT_VT) {
          const int bb = gm >> 10, ss = gm & 1023;
          const int hh = gn >> 6,  hd = gn & 63;
          ((bf16*)Out)[(((size_t)(bb * H + hh)) * HD + hd) * S + ss] = f2bf(val);
        } else {
          ((float*)Out)[(size_t)gm * D + gn] = val;
        }
      }
}

// ---------- causal flash attention ----------
// One wave per (16-query tile, b*h). Q kept as 2 A-frags. Per 32-key tile:
// 4 WMMAs for S = Q K^T, online softmax (16-lane row reductions),
// P -> LDS -> A-frag, 4 WMMAs for O += P V (V pre-transposed per head).

__global__ __launch_bounds__(32)
void flash_attn(const bf16* __restrict__ Qh, const bf16* __restrict__ Kh,
                const bf16* __restrict__ Vt, bf16* __restrict__ Oh) {
  constexpr int LDP = 40;
  __shared__ bf16 Ps[16 * LDP];

  const int lane = threadIdx.x;
  const int qt = blockIdx.x;          // 0..S/16-1
  const int bh = blockIdx.y;          // 0..B*H-1
  const int b = bh >> 4, h = bh & 15;
  const int qbase = qt * 16;
  const int mA = lane & 15, kh = lane >> 4;

  // Q fragments (16x64 -> 2 x A[16x32])
  const bf16* qrow = Qh + ((size_t)(b * S + qbase + mA) * D + h * HD);
  v16bf aq[2];
#pragma unroll
  for (int f = 0; f < 2; ++f)
    aq[f] = ld_frag(qrow + f * 32 + kh * 16, qrow + f * 32 + kh * 16 + 8);

  float m_run[8], l_run[8];
  v8f o[4];
  const v8f vz = {0.f, 0.f, 0.f, 0.f, 0.f, 0.f, 0.f, 0.f};
#pragma unroll
  for (int ni = 0; ni < 4; ++ni) o[ni] = vz;
#pragma unroll
  for (int r = 0; r < 8; ++r) { m_run[r] = -1e30f; l_run[r] = 0.f; }

  const float scale = 0.125f;  // 1/sqrt(64)
  const int ntile = (qbase + 15) / 32;  // last key tile index (causal)

  for (int jt = 0; jt <= ntile; ++jt) {
    const int j0 = jt * 32;
    if (jt + 1 <= ntile) {  // prefetch next K/V tiles into near cache
      __builtin_prefetch(Kh + ((size_t)(b * S + j0 + 32 + mA) * D + h * HD), 0, 3);
      __builtin_prefetch(Vt + ((size_t)(bh * HD + mA) * S + j0 + 32), 0, 3);
    }
    // ---- scores: two 16x16 C-tiles, contraction over hd=64 in two steps ----
    v8f c[2];
#pragma unroll
    for (int t = 0; t < 2; ++t) {
      const bf16* krow =
          Kh + ((size_t)(b * S + j0 + t * 16 + mA) * D + h * HD + kh * 16);
      v16bf bk0 = ld_frag(krow,      krow + 8);       // hd 0..31
      v16bf bk1 = ld_frag(krow + 32, krow + 40);      // hd 32..63
      v8f cc = vz;
      cc = wmma_bf16(aq[0], bk0, cc);
      cc = wmma_bf16(aq[1], bk1, cc);
      c[t] = cc;
    }

    const bool needmask = (j0 + 31) > qbase;
#pragma unroll
    for (int r = 0; r < 8; ++r) {
      const int qidx = qbase + r + 8 * kh;
      if (needmask) {
        c[0][r] = (j0 + mA      > qidx) ? -1e30f : c[0][r] * scale;
        c[1][r] = (j0 + 16 + mA > qidx) ? -1e30f : c[1][r] * scale;
      } else {
        c[0][r] *= scale;
        c[1][r] *= scale;
      }
    }

    // ---- online softmax (row = 16-lane group within a VGPR) ----
    float alpha[8], p0[8], p1[8];
#pragma unroll
    for (int r = 0; r < 8; ++r) {
      float t = fmaxf(c[0][r], c[1][r]);
#pragma unroll
      for (int off = 1; off < 16; off <<= 1)
        t = fmaxf(t, __shfl_xor(t, off, 32));
      const float mn = fmaxf(m_run[r], t);
      const float a  = __expf(m_run[r] - mn);
      const float e0 = __expf(c[0][r] - mn);
      const float e1 = __expf(c[1][r] - mn);
      float ls = e0 + e1;
#pragma unroll
      for (int off = 1; off < 16; off <<= 1)
        ls += __shfl_xor(ls, off, 32);
      m_run[r] = mn;
      l_run[r] = l_run[r] * a + ls;
      alpha[r] = a; p0[r] = e0; p1[r] = e1;
    }
#pragma unroll
    for (int ni = 0; ni < 4; ++ni)
#pragma unroll
      for (int r = 0; r < 8; ++r) o[ni][r] *= alpha[r];

    // ---- P: C layout -> LDS -> A-fragment (16x32 bf16) ----
#pragma unroll
    for (int r = 0; r < 8; ++r) {
      const int M = r + 8 * kh;
      Ps[M * LDP + mA]      = f2bf(p0[r]);
      Ps[M * LDP + 16 + mA] = f2bf(p1[r]);
    }
    __syncthreads();
    const v16bf ap = ld_frag(&Ps[mA * LDP + kh * 16], &Ps[mA * LDP + kh * 16 + 8]);

    // ---- O += P * V  (V transposed: rows contiguous along keys) ----
#pragma unroll
    for (int ni = 0; ni < 4; ++ni) {
      const bf16* vrow =
          Vt + ((size_t)(bh * HD + ni * 16 + mA) * S + j0 + kh * 16);
      v16bf bv = ld_frag(vrow, vrow + 8);
      o[ni] = wmma_bf16(ap, bv, o[ni]);
    }
    __syncthreads();
  }

  // ---- normalize and write bf16 attention output [token, h*64+hd] ----
#pragma unroll
  for (int ni = 0; ni < 4; ++ni)
#pragma unroll
    for (int r = 0; r < 8; ++r) {
      const int M = r + 8 * kh;
      const float val = o[ni][r] / l_run[r];
      Oh[(size_t)(b * S + qbase + M) * D + h * HD + ni * 16 + mA] = f2bf(val);
    }
}

// ---------- launcher ----------

extern "C" void kernel_launch(void* const* d_in, const int* in_sizes, int n_in,
                              void* d_out, int out_size, void* d_ws, size_t ws_size,
                              hipStream_t stream) {
  (void)in_sizes; (void)n_in; (void)out_size; (void)ws_size;
  const float* q  = (const float*)d_in[0];
  const float* k  = (const float*)d_in[1];
  const float* v  = (const float*)d_in[2];
  const float* Wq = (const float*)d_in[3];
  const float* Wk = (const float*)d_in[4];
  const float* Wv = (const float*)d_in[5];
  const float* Wo = (const float*)d_in[6];
  float* out = (float*)d_out;

  const size_t NTOK = (size_t)M_TOK * D;  // 8Mi elems
  const size_t NW   = (size_t)D * D;      // 1Mi elems
  bf16* qb  = (bf16*)d_ws;        // bf16 casts
  bf16* kb  = qb  + NTOK;
  bf16* vb  = kb  + NTOK;
  bf16* wqb = vb  + NTOK;
  bf16* wkb = wqb + NW;
  bf16* wvb = wkb + NW;
  bf16* wob = wvb + NW;
  bf16* Qh  = wob + NW;           // [8192,1024] bf16
  bf16* Kh  = Qh  + NTOK;         // [8192,1024] bf16
  bf16* Vt  = Kh  + NTOK;         // [B*H,64,S]  bf16
  bf16* Ah  = qb;                 // reuse: q-bf16 dead after Q projection

  const dim3 cb(256);
  cast_f32_bf16<<<dim3((int)(NTOK / 8 / 256)), cb, 0, stream>>>(q,  qb,  (int)NTOK);
  cast_f32_bf16<<<dim3((int)(NTOK / 8 / 256)), cb, 0, stream>>>(k,  kb,  (int)NTOK);
  cast_f32_bf16<<<dim3((int)(NTOK / 8 / 256)), cb, 0, stream>>>(v,  vb,  (int)NTOK);
  cast_f32_bf16<<<dim3((int)(NW   / 8 / 256)), cb, 0, stream>>>(Wq, wqb, (int)NW);
  cast_f32_bf16<<<dim3((int)(NW   / 8 / 256)), cb, 0, stream>>>(Wk, wkb, (int)NW);
  cast_f32_bf16<<<dim3((int)(NW   / 8 / 256)), cb, 0, stream>>>(Wv, wvb, (int)NW);
  cast_f32_bf16<<<dim3((int)(NW   / 8 / 256)), cb, 0, stream>>>(Wo, wob, (int)NW);

  const dim3 gg(D / 256, M_TOK / 128), gb(256);
  gemm_bf16<OUT_BF16><<<gg, gb, 0, stream>>>(qb, wqb, Qh);
  gemm_bf16<OUT_BF16><<<gg, gb, 0, stream>>>(kb, wkb, Kh);
  gemm_bf16<OUT_VT  ><<<gg, gb, 0, stream>>>(vb, wvb, Vt);
  flash_attn<<<dim3(S / 16, Bb * H), dim3(32), 0, stream>>>(Qh, Kh, Vt, Ah);
  gemm_bf16<OUT_F32 ><<<gg, gb, 0, stream>>>(Ah, wob, out);
}
